// paramnet_83408264888942
// MI455X (gfx1250) — matrix operand order repaired and checked by reading, not compiled, not run on hardware
//
#include <hip/hip_runtime.h>
#include <cstddef>

typedef __attribute__((ext_vector_type(16))) _Float16 v16h;
typedef __attribute__((ext_vector_type(8)))  _Float16 v8h;
typedef __attribute__((ext_vector_type(8)))  float    v8f;

static int div_up_h(int a, int b) { return (a + b - 1) / b; }
static int r16(int a) { return (a + 15) & ~15; }
static int r32(int a) { return (a + 31) & ~31; }

// ---------------------------------------------------------------------------
// kNN (k=100) + neighbor gather fused. One block = 128 queries of one batch
// (1024 % 128 == 0), cloud xyz staged in LDS once and re-scanned from there.
// ---------------------------------------------------------------------------
__global__ __launch_bounds__(128) void knn_gather_kernel(
    const float* __restrict__ pts, float* __restrict__ flat, int B, int N, int K) {
  __shared__ float sP[1024 * 3];
  int tid = blockIdx.x * blockDim.x + threadIdx.x;
  int b = (blockIdx.x * (int)blockDim.x) / N;  // uniform per block
  const float* P = pts + (size_t)b * N * 3;
  for (int j = threadIdx.x; j < N * 3; j += blockDim.x) sP[j] = P[j];
  __syncthreads();
  if (tid >= B * N) return;
  int i = tid - b * N;
  float xi = sP[i * 3 + 0], yi = sP[i * 3 + 1], zi = sP[i * 3 + 2];
  float bd[100]; int bi[100];
  for (int k = 0; k < 100; ++k) { bd[k] = 3.0e38f; bi[k] = 0; }
  for (int j = 0; j < N; ++j) {
    float dx = sP[j * 3 + 0] - xi, dy = sP[j * 3 + 1] - yi, dz = sP[j * 3 + 2] - zi;
    float d = dx * dx + dy * dy + dz * dz;
    if (d < bd[K - 1]) {
      int p = K - 1;
      while (p > 0 && bd[p - 1] > d) { bd[p] = bd[p - 1]; bi[p] = bi[p - 1]; --p; }
      bd[p] = d; bi[p] = j;
    }
  }
  float* o = flat + (size_t)tid * K * 3;
  for (int k = 0; k < K; ++k) {
    int j = bi[k];
    o[k * 3 + 0] = sP[j * 3 + 0];
    o[k * 3 + 1] = sP[j * 3 + 1];
    o[k * 3 + 2] = sP[j * 3 + 2];
  }
}

// ---------------------------------------------------------------------------
// Farthest point sampling: one block (128 thr) per cloud, N <= 1024.
// Cloud xyz staged in LDS (re-scanned S times).
// ---------------------------------------------------------------------------
__global__ __launch_bounds__(128) void fps_kernel(const float* __restrict__ xyz,
                                                  int* __restrict__ fidx,
                                                  int N, int S) {
  int c = blockIdx.x;
  const float* P = xyz + (size_t)c * N * 3;
  int* F = fidx + (size_t)c * S;
  __shared__ float sP[1024 * 3];
  __shared__ float dist[1024];
  __shared__ float sd[128];
  __shared__ int   si[128];
  __shared__ int   slast;
  int t = threadIdx.x;
  for (int j = t; j < N * 3; j += blockDim.x) sP[j] = P[j];
  for (int j = t; j < N; j += blockDim.x) dist[j] = 1e10f;
  if (t == 0) { F[0] = 0; slast = 0; }
  __syncthreads();
  for (int s = 1; s < S; ++s) {
    int last = slast;
    float lx = sP[last * 3], ly = sP[last * 3 + 1], lz = sP[last * 3 + 2];
    float bm = -1.0f; int bj = 0;
    for (int j = t; j < N; j += blockDim.x) {
      float dx = sP[j * 3] - lx, dy = sP[j * 3 + 1] - ly, dz = sP[j * 3 + 2] - lz;
      float d = dx * dx + dy * dy + dz * dz;
      float nd = fminf(dist[j], d);
      dist[j] = nd;
      if (nd > bm) { bm = nd; bj = j; }
    }
    sd[t] = bm; si[t] = bj;
    __syncthreads();
    for (int off = 64; off > 0; off >>= 1) {
      if (t < off) {
        if (sd[t + off] > sd[t] ||
            (sd[t + off] == sd[t] && si[t + off] < si[t])) {
          sd[t] = sd[t + off]; si[t] = si[t + off];
        }
      }
      __syncthreads();
    }
    if (t == 0) { F[s] = si[0]; slast = si[0]; }
    __syncthreads();
  }
}

__global__ void gather_centers_kernel(const float* __restrict__ xyz,
                                      const int* __restrict__ fidx,
                                      float* __restrict__ nxyz,
                                      int ncloud, int N, int S) {
  int tid = blockIdx.x * blockDim.x + threadIdx.x;
  if (tid >= ncloud * S) return;
  int c = tid / S;
  int j = fidx[tid];
  const float* p = xyz + ((size_t)c * N + j) * 3;
  float* o = nxyz + (size_t)tid * 3;
  o[0] = p[0]; o[1] = p[1]; o[2] = p[2];
}

__global__ void ball_query_kernel(const float* __restrict__ xyz,
                                  const float* __restrict__ nxyz,
                                  int* __restrict__ bidx,
                                  int ncloud, int N, int S, int ns, float r2) {
  int tid = blockIdx.x * blockDim.x + threadIdx.x;
  if (tid >= ncloud * S) return;
  int c = tid / S;
  const float* q = nxyz + (size_t)tid * 3;
  const float* P = xyz + (size_t)c * N * 3;
  int* o = bidx + (size_t)tid * ns;
  int cnt = 0, first = N - 1;
  for (int j = 0; j < N && cnt < ns; ++j) {
    float dx = P[j * 3] - q[0], dy = P[j * 3 + 1] - q[1], dz = P[j * 3 + 2] - q[2];
    if (dx * dx + dy * dy + dz * dz <= r2) {
      if (cnt == 0) first = j;
      o[cnt++] = j;
    }
  }
  for (int k = cnt; k < ns; ++k) o[k] = first;
}

// ---------------------------------------------------------------------------
// Producers of padded f16 GEMM inputs (rows Mp x Cp, zero-padded)
// ---------------------------------------------------------------------------
__global__ void group_h_kernel(const float* __restrict__ xyz,
                               const float* __restrict__ feats,
                               const float* __restrict__ nxyz,
                               const int* __restrict__ bidx,
                               _Float16* __restrict__ g,
                               int ncloud, int N, int S, int ns,
                               int Cf, int fstride, int Cp, int totalp) {
  int tid = blockIdx.x * blockDim.x + threadIdx.x;
  if (tid >= totalp) return;
  int total = ncloud * S * ns;
  _Float16* o = g + (size_t)tid * Cp;
  if (tid >= total) {
    for (int ch = 0; ch < Cp; ++ch) o[ch] = (_Float16)0.0f;
    return;
  }
  int ps = tid / ns;  // c*S + s
  int c = ps / S;
  int j = bidx[tid];
  const float* p = xyz + ((size_t)c * N + j) * 3;
  const float* q = nxyz + (size_t)ps * 3;
  o[0] = (_Float16)(p[0] - q[0]);
  o[1] = (_Float16)(p[1] - q[1]);
  o[2] = (_Float16)(p[2] - q[2]);
  if (Cf > 0) {
    const float* f = feats + ((size_t)c * N + j) * fstride;
    for (int ch = 0; ch < Cf; ++ch) o[3 + ch] = (_Float16)f[ch];
  }
  for (int ch = 3 + Cf; ch < Cp; ++ch) o[ch] = (_Float16)0.0f;
}

__global__ void concat_all_h_kernel(const float* __restrict__ xyz,
                                    const float* __restrict__ feats,
                                    _Float16* __restrict__ g,
                                    int total, int Cf, int fstride, int Cp, int totalp) {
  int tid = blockIdx.x * blockDim.x + threadIdx.x;
  if (tid >= totalp) return;
  _Float16* o = g + (size_t)tid * Cp;
  if (tid >= total) {
    for (int ch = 0; ch < Cp; ++ch) o[ch] = (_Float16)0.0f;
    return;
  }
  const float* p = xyz + (size_t)tid * 3;
  o[0] = (_Float16)p[0]; o[1] = (_Float16)p[1]; o[2] = (_Float16)p[2];
  const float* f = feats + (size_t)tid * fstride;
  for (int ch = 0; ch < Cf; ++ch) o[3 + ch] = (_Float16)f[ch];
  for (int ch = 3 + Cf; ch < Cp; ++ch) o[ch] = (_Float16)0.0f;
}

// f32 (M,K) -> f16 (Mp,Kp) zero padded
__global__ void cvtpad_kernel(const float* __restrict__ in, _Float16* __restrict__ out,
                              int M, int K, int Kp, int Mp) {
  int tid = blockIdx.x * blockDim.x + threadIdx.x;
  if (tid >= Mp * Kp) return;
  int m = tid / Kp, k = tid - m * Kp;
  float v = (m < M && k < K) ? in[(size_t)m * K + k] : 0.0f;
  out[tid] = (_Float16)v;
}

// weight prep: W (K,N) f32 -> Wt (Np,Kp) f16 transposed+padded ; bias -> bp (Np)
__global__ void wprep_kernel(const float* __restrict__ W, const float* __restrict__ b,
                             _Float16* __restrict__ Wt, float* __restrict__ bp,
                             int K, int N, int Kp, int Np) {
  int tid = blockIdx.x * blockDim.x + threadIdx.x;
  int total = Np * Kp;
  if (tid < total) {
    int n = tid / Kp, k = tid - n * Kp;
    float v = (k < K && n < N) ? W[(size_t)k * N + n] : 0.0f;
    Wt[tid] = (_Float16)v;
  }
  if (tid < Np) bp[tid] = (tid < N) ? b[tid] : 0.0f;
}

// max over sample dimension: out[g, ch] = max_k in[(g*ns+k), ch]   (f32)
__global__ void rowmax_kernel(const float* __restrict__ in, float* __restrict__ out,
                              int ngroups, int ns, int C) {
  int tid = blockIdx.x * blockDim.x + threadIdx.x;
  if (tid >= ngroups * C) return;
  int g = tid / C, ch = tid - g * C;
  const float* p = in + (size_t)g * ns * C + ch;
  float m = p[0];
  for (int k = 1; k < ns; ++k) m = fmaxf(m, p[(size_t)k * C]);
  out[tid] = m;
}

// ---------------------------------------------------------------------------
// WMMA GEMM, fully padded f16 operands, branch-free software-pipelined loop.
//   A  : (Mp x Kp) f16 row-major, Kp % 32 == 0
//   Wt : (Np x Kp) f16 = W^T padded (Np % 32 == 0)
// Each wave owns a 16x32 output tile: one A fragment feeds two WMMAs
// (cols col0..col0+15 and col0+16..col0+31), double-buffered loads so the
// WMMA of step i overlaps the 6x b128 loads of step i+1.
// ---------------------------------------------------------------------------
__global__ __launch_bounds__(256) void wmma_gemm_kernel(
    const _Float16* __restrict__ A, const _Float16* __restrict__ Wt,
    const float* __restrict__ bias, float* __restrict__ Cf,
    _Float16* __restrict__ Ch, int M, int N, int Np, int Kp,
    int tiles_m, int relu) {
  int wave = threadIdx.x >> 5;
  int lane = threadIdx.x & 31;
  int half = lane >> 4;
  int idx16 = lane & 15;
  int tiles_n = Np >> 5;  // 32-wide output tiles
  int tile_id = blockIdx.x * 8 + wave;
  int tile_m = tile_id / tiles_n;
  int tile_n = tile_id - tile_m * tiles_n;
  if (tile_m >= tiles_m) { tile_m = tiles_m - 1; tile_n = 0; }  // dup tile, benign
  int row0 = tile_m * 16, col0 = tile_n * 32;

  // A frag: lane<16 -> M=lane, K k0..k0+15; lane>=16 shifted by 16
  const _Float16* arow  = A + (size_t)(row0 + idx16) * Kp + half * 16;
  // B frags from W^T: lanes 0-15 hold K k0..k0+15, lanes 16-31 hold +16
  const _Float16* brow0 = Wt + (size_t)(col0 + idx16) * Kp + half * 16;
  const _Float16* brow1 = brow0 + (size_t)16 * Kp;

  v8f acc0 = {}, acc1 = {};
  v8h a0 = *(const v8h*)(arow);
  v8h a1 = *(const v8h*)(arow + 8);
  v16h b0 = *(const v16h*)(brow0);
  v16h b1 = *(const v16h*)(brow1);

  for (int k0 = 0; k0 < Kp; k0 += 32) {
    int kn = (k0 + 32 < Kp) ? (k0 + 32) : 0;  // clamped: always-valid prefetch slot
    v8h na0 = *(const v8h*)(arow + kn);
    v8h na1 = *(const v8h*)(arow + kn + 8);
    v16h nb0 = *(const v16h*)(brow0 + kn);
    v16h nb1 = *(const v16h*)(brow1 + kn);
    if (k0 + 64 < Kp) {
      __builtin_prefetch(arow + k0 + 64, 0, 3);   // global_prefetch_b8
      __builtin_prefetch(brow0 + k0 + 64, 0, 3);
      __builtin_prefetch(brow1 + k0 + 64, 0, 3);
    }
    v16h af = __builtin_shufflevector(a0, a1, 0, 1, 2, 3, 4, 5, 6, 7,
                                      8, 9, 10, 11, 12, 13, 14, 15);
    acc0 = __builtin_amdgcn_wmma_f32_16x16x32_f16(
        false, af, false, b0, (short)0, acc0, false, false);
    acc1 = __builtin_amdgcn_wmma_f32_16x16x32_f16(
        false, af, false, b1, (short)0, acc1, false, false);
    a0 = na0; a1 = na1; b0 = nb0; b1 = nb1;
  }

  // epilogue: bias + optional relu; C/D layout VGPR r -> M=r+8*half, N=idx16
  int gn0 = col0 + idx16;
  int gn1 = col0 + 16 + idx16;
#pragma unroll
  for (int r = 0; r < 8; ++r) {
    int gm = row0 + r + half * 8;
    float v0 = acc0[r] + bias[gn0];
    float v1 = acc1[r] + bias[gn1];
    if (relu) { v0 = fmaxf(v0, 0.0f); v1 = fmaxf(v1, 0.0f); }
    if (Ch) {
      Ch[(size_t)gm * Np + gn0] = (_Float16)v0;  // padded, unguarded
      Ch[(size_t)gm * Np + gn1] = (_Float16)v1;
    } else {
      if (gm < M && gn0 < N) Cf[(size_t)gm * N + gn0] = v0;
      if (gm < M && gn1 < N) Cf[(size_t)gm * N + gn1] = v1;
    }
  }
}

// attr = [points(3), eula(3), softmax(edge)(2), softmax(meta)(7)]
__global__ void attr_kernel(const float* __restrict__ pts,
                            const float* __restrict__ eula,
                            const float* __restrict__ edge,
                            const float* __restrict__ meta,
                            float* __restrict__ attr, int BN) {
  int tid = blockIdx.x * blockDim.x + threadIdx.x;
  if (tid >= BN) return;
  float* o = attr + (size_t)tid * 15;
  o[0] = pts[tid * 3 + 0]; o[1] = pts[tid * 3 + 1]; o[2] = pts[tid * 3 + 2];
  o[3] = eula[tid * 3 + 0]; o[4] = eula[tid * 3 + 1]; o[5] = eula[tid * 3 + 2];
  float e0 = edge[tid * 2 + 0], e1 = edge[tid * 2 + 1];
  float em = fmaxf(e0, e1);
  float x0 = expf(e0 - em), x1 = expf(e1 - em);
  float es = x0 + x1;
  o[6] = x0 / es; o[7] = x1 / es;
  float mm = -3.0e38f;
  for (int i = 0; i < 7; ++i) mm = fmaxf(mm, meta[tid * 7 + i]);
  float mv[7], s = 0.0f;
  for (int i = 0; i < 7; ++i) { mv[i] = expf(meta[tid * 7 + i] - mm); s += mv[i]; }
  for (int i = 0; i < 7; ++i) o[8 + i] = mv[i] / s;
}

__global__ void extract3_kernel(const float* __restrict__ src, float* __restrict__ dst,
                                int n, int stride) {
  int tid = blockIdx.x * blockDim.x + threadIdx.x;
  if (tid >= n) return;
  dst[tid * 3 + 0] = src[(size_t)tid * stride + 0];
  dst[tid * 3 + 1] = src[(size_t)tid * stride + 1];
  dst[tid * 3 + 2] = src[(size_t)tid * stride + 2];
}

__global__ void logsoftmax_kernel(const float* __restrict__ in, float* __restrict__ out,
                                  int rows, int C) {
  int r = blockIdx.x * blockDim.x + threadIdx.x;
  if (r >= rows) return;
  const float* x = in + (size_t)r * C;
  float* o = out + (size_t)r * C;
  float m = x[0];
  for (int c = 1; c < C; ++c) m = fmaxf(m, x[c]);
  float s = 0.0f;
  for (int c = 0; c < C; ++c) s += expf(x[c] - m);
  float ls = logf(s);
  for (int c = 0; c < C; ++c) o[c] = x[c] - m - ls;
}

// ---------------------------------------------------------------------------
extern "C" void kernel_launch(void* const* d_in, const int* in_sizes, int n_in,
                              void* d_out, int out_size, void* d_ws, size_t ws_size,
                              hipStream_t stream) {
  (void)in_sizes; (void)n_in; (void)out_size; (void)ws_size;

  const float* P = (const float*)d_in[0];  // (2, 1024, 3)

  // ----- raw parameter pointers (setup_inputs dict insertion order) -----
  int pi = 1;
  const float *bb0W[3][3], *bb0B[3][3], *bb1W[3][3], *bb1B[3][3];
  for (int l = 0; l < 3; ++l) {
    for (int j = 0; j < 3; ++j) bb0W[l][j] = (const float*)d_in[pi + j];
    for (int j = 0; j < 3; ++j) bb0B[l][j] = (const float*)d_in[pi + 3 + j];
    pi += 6;
  }
  for (int l = 0; l < 3; ++l) {
    for (int j = 0; j < 3; ++j) bb1W[l][j] = (const float*)d_in[pi + j];
    for (int j = 0; j < 3; ++j) bb1B[l][j] = (const float*)d_in[pi + 3 + j];
    pi += 6;
  }
  const float *eulaW[4], *eulaB[4], *edgeW[4], *edgeB[4], *metaW[4], *metaB[4];
  for (int j = 0; j < 4; ++j) eulaW[j] = (const float*)d_in[pi + j];
  for (int j = 0; j < 4; ++j) eulaB[j] = (const float*)d_in[pi + 4 + j];
  pi += 8;
  for (int j = 0; j < 4; ++j) edgeW[j] = (const float*)d_in[pi + j];
  for (int j = 0; j < 4; ++j) edgeB[j] = (const float*)d_in[pi + 4 + j];
  pi += 8;
  for (int j = 0; j < 4; ++j) metaW[j] = (const float*)d_in[pi + j];
  for (int j = 0; j < 4; ++j) metaB[j] = (const float*)d_in[pi + 4 + j];
  pi += 8;
  const float *finW[5], *finB[5];
  for (int j = 0; j < 5; ++j) finW[j] = (const float*)d_in[pi + j];
  for (int j = 0; j < 5; ++j) finB[j] = (const float*)d_in[pi + 5 + j];

  // ----- workspace (32B-aligned slices) -----
  float* ws = (float*)d_ws;
  size_t off = 0;
  auto fall = [&](size_t n) {
    float* p = ws + off;
    off += (n + 7) & ~(size_t)7;  // keep 32B alignment
    return p;
  };

  float* flat    = fall((size_t)2048 * 100 * 3);
  int*   fidx1   = (int*)fall((size_t)2048 * 50);
  float* nxyz1   = fall((size_t)2048 * 50 * 3);
  int*   bidx1   = (int*)fall((size_t)2048 * 50 * 9);
  int*   fidx2   = (int*)fall((size_t)2048 * 25);
  float* nxyz2   = fall((size_t)2048 * 25 * 3);
  int*   bidx2   = (int*)fall((size_t)2048 * 25 * 12);
  float* feats1  = fall((size_t)2048 * 50 * 32);
  float* feats2  = fall((size_t)2048 * 25 * 64);
  float* ex      = fall((size_t)2048 * 256);
  float* attr    = fall((size_t)2 * 1024 * 15);
  float* xyzB    = fall((size_t)2 * 1024 * 3);
  int*   fidxB1  = (int*)fall((size_t)2 * 512);
  float* nxyzB1  = fall((size_t)2 * 512 * 3);
  int*   bidxB1  = (int*)fall((size_t)2 * 512 * 32);
  float* featsB1 = fall((size_t)2 * 512 * 128);
  int*   fidxB2  = (int*)fall((size_t)2 * 128);
  float* nxyzB2  = fall((size_t)2 * 128 * 3);
  int*   bidxB2  = (int*)fall((size_t)2 * 128 * 64);
  float* featsB2 = fall((size_t)2 * 128 * 256);
  float* gfeat   = fall((size_t)2 * 1024);
  float* logits  = fall(16);
  _Float16* exh  = (_Float16*)fall((size_t)2048 * 256 / 2);
  _Float16* gfh  = (_Float16*)fall((size_t)16 * 1024 / 2);
  _Float16* PAh  = (_Float16*)fall((size_t)20000000);  // 40M halves
  _Float16* PBh  = (_Float16*)fall((size_t)15000000);  // 30M halves
  float*    PF   = fall((size_t)40000000);             // 40M floats

  float* out      = (float*)d_out;
  float* out_log  = out;                    // (2,2)
  float* out_eula = out + 4;                // (2,1024,3)
  float* out_edge = out + 4 + 6144;         // (2,1024,2)
  float* out_meta = out + 4 + 6144 + 4096;  // (2,1024,7)

  // ----- per-layer prepped weights -----
  struct Layer { int K, N, Kp, Np; _Float16* Wt; float* bp; };
  auto mkL = [&](const float* W, const float* B, int K, int N) {
    Layer L;
    L.K = K; L.N = N; L.Kp = r32(K); L.Np = r32(N);
    L.Wt = (_Float16*)fall((size_t)L.Np * L.Kp / 2);
    L.bp = fall(L.Np);
    int total = L.Np * L.Kp;
    wprep_kernel<<<div_up_h(total, 256), 256, 0, stream>>>(W, B, L.Wt, L.bp, K, N, L.Kp, L.Np);
    return L;
  };

  Layer Lb0[3][3], Lb1[3][3], Le[4], Ld[4], Lm[4], Lf[5];
  const int D0[3][4] = {{3, 16, 16, 32}, {35, 32, 32, 64}, {67, 64, 128, 256}};
  const int D1[3][4] = {{15, 64, 64, 128}, {131, 128, 128, 256}, {259, 256, 512, 1024}};
  for (int l = 0; l < 3; ++l)
    for (int j = 0; j < 3; ++j) Lb0[l][j] = mkL(bb0W[l][j], bb0B[l][j], D0[l][j], D0[l][j + 1]);
  for (int l = 0; l < 3; ++l)
    for (int j = 0; j < 3; ++j) Lb1[l][j] = mkL(bb1W[l][j], bb1B[l][j], D1[l][j], D1[l][j + 1]);
  const int DE[5] = {256, 256, 128, 32, 3};
  const int DD[5] = {256, 256, 128, 32, 2};
  const int DM[5] = {256, 256, 128, 32, 7};
  const int DF[6] = {1024, 512, 256, 128, 64, 2};
  for (int j = 0; j < 4; ++j) Le[j] = mkL(eulaW[j], eulaB[j], DE[j], DE[j + 1]);
  for (int j = 0; j < 4; ++j) Ld[j] = mkL(edgeW[j], edgeB[j], DD[j], DD[j + 1]);
  for (int j = 0; j < 4; ++j) Lm[j] = mkL(metaW[j], metaB[j], DM[j], DM[j + 1]);
  for (int j = 0; j < 5; ++j) Lf[j] = mkL(finW[j], finB[j], DF[j], DF[j + 1]);

  // ----- helpers -----
  auto gemm = [&](const _Float16* A, const Layer& L, _Float16* oh, float* of,
                  int M, int relu) {
    int tiles_m = r16(M) >> 4;
    int tiles = tiles_m * (L.Np >> 5);
    wmma_gemm_kernel<<<div_up_h(tiles, 8), 256, 0, stream>>>(
        A, L.Wt, L.bp, of, oh, M, L.N, L.Np, L.Kp, tiles_m, relu);
  };

  auto run_sa = [&](const float* xyz, const float* feats, int Cf, int fstride,
                    int ncloud, int N, int S, int ns, float radius,
                    const Layer* L, int* fidx, float* nxyz, int* bidx,
                    float* outfeats) {
    fps_kernel<<<ncloud, 128, 0, stream>>>(xyz, fidx, N, S);
    int nS = ncloud * S;
    gather_centers_kernel<<<div_up_h(nS, 256), 256, 0, stream>>>(xyz, fidx, nxyz, ncloud, N, S);
    ball_query_kernel<<<div_up_h(nS, 256), 256, 0, stream>>>(xyz, nxyz, bidx, ncloud, N, S, ns,
                                                             radius * radius);
    int rows = nS * ns;
    int rowsp = r16(rows);
    group_h_kernel<<<div_up_h(rowsp, 256), 256, 0, stream>>>(
        xyz, feats, nxyz, bidx, PAh, ncloud, N, S, ns, Cf, fstride, L[0].Kp, rowsp);
    gemm(PAh, L[0], PBh, nullptr, rows, 1);
    gemm(PBh, L[1], PAh, nullptr, rows, 1);
    gemm(PAh, L[2], nullptr, PF, rows, 1);
    rowmax_kernel<<<div_up_h(nS * L[2].N, 256), 256, 0, stream>>>(PF, outfeats, nS, ns, L[2].N);
  };

  auto run_sa_last = [&](const float* xyz, const float* feats, int Cf, int fstride,
                         int ncloud, int N, const Layer* L, float* outfeats) {
    int rows = ncloud * N;
    int rowsp = r16(rows);
    concat_all_h_kernel<<<div_up_h(rowsp, 256), 256, 0, stream>>>(
        xyz, feats, PAh, rows, Cf, fstride, L[0].Kp, rowsp);
    gemm(PAh, L[0], PBh, nullptr, rows, 1);
    gemm(PBh, L[1], PAh, nullptr, rows, 1);
    gemm(PAh, L[2], nullptr, PF, rows, 1);
    rowmax_kernel<<<div_up_h(ncloud * L[2].N, 256), 256, 0, stream>>>(PF, outfeats, ncloud, N,
                                                                      L[2].N);
  };

  auto run_mlp = [&](const _Float16* inh, const Layer* L, int nl, int M, float* dst_f32) {
    const _Float16* cur = inh;
    for (int i = 0; i < nl; ++i) {
      if (i == nl - 1) {
        gemm(cur, L[i], nullptr, dst_f32, M, 0);
      } else {
        _Float16* dst = (i % 2 == 0) ? PAh : PBh;
        gemm(cur, L[i], dst, nullptr, M, 1);
        cur = dst;
      }
    }
  };

  // ----- pipeline -----
  knn_gather_kernel<<<div_up_h(2 * 1024, 128), 128, 0, stream>>>(P, flat, 2, 1024, 100);

  // bb0: 2048 clouds of 100 neighbor points
  run_sa(flat, nullptr, 0, 0, 2048, 100, 50, 9, 0.2f, Lb0[0], fidx1, nxyz1, bidx1, feats1);
  run_sa(nxyz1, feats1, 32, 32, 2048, 50, 25, 12, 0.4f, Lb0[1], fidx2, nxyz2, bidx2, feats2);
  run_sa_last(nxyz2, feats2, 64, 64, 2048, 25, Lb0[2], ex);

  // heads on ex (2048, 256) — raw logits straight into d_out
  cvtpad_kernel<<<div_up_h(2048 * 256, 256), 256, 0, stream>>>(ex, exh, 2048, 256, 256, 2048);
  run_mlp(exh, Le, 4, 2048, out_eula);
  run_mlp(exh, Ld, 4, 2048, out_edge);
  run_mlp(exh, Lm, 4, 2048, out_meta);

  // attr = [pts, eula, softmax(edge), softmax(meta)]  (2*1024, 15)
  attr_kernel<<<div_up_h(2048, 256), 256, 0, stream>>>(P, out_eula, out_edge, out_meta, attr,
                                                       2048);
  extract3_kernel<<<div_up_h(2048, 256), 256, 0, stream>>>(attr, xyzB, 2048, 15);

  // bb1
  run_sa(xyzB, attr + 3, 12, 15, 2, 1024, 512, 32, 0.2f, Lb1[0], fidxB1, nxyzB1, bidxB1,
         featsB1);
  run_sa(nxyzB1, featsB1, 128, 128, 2, 512, 128, 64, 0.4f, Lb1[1], fidxB2, nxyzB2, bidxB2,
         featsB2);
  run_sa_last(nxyzB2, featsB2, 256, 256, 2, 128, Lb1[2], gfeat);

  // final classifier + log_softmax
  cvtpad_kernel<<<div_up_h(16 * 1024, 256), 256, 0, stream>>>(gfeat, gfh, 2, 1024, 1024, 16);
  run_mlp(gfh, Lf, 5, 2, logits);
  logsoftmax_kernel<<<1, 32, 0, stream>>>(logits, out_log, 2, 2);
}
